// RPNGTMatcher_23931557773456
// MI455X (gfx1250) — compile-verified, matrix operand order checked by reading, not run on hardware
//
#include <hip/hip_runtime.h>
#include <hip/hip_bf16.h>
#include <stdint.h>

// ---------------- problem constants (match reference) ----------------
#define N_B 4
#define N_G 64
#define POS_THR 0.5f
#define NEG_THR 0.3f
#define SAMPLE_BATCH 512
#define TARGET_FG 153
#define CAP 1024   // max collected provisional positives per image (>> TARGET_FG)

typedef __attribute__((ext_vector_type(16))) _Float16 v16h;
typedef __attribute__((ext_vector_type(8)))  float    v8f;
typedef unsigned int v4u __attribute__((ext_vector_type(4)));
typedef int          v8i __attribute__((ext_vector_type(8)));
typedef int          v4i __attribute__((ext_vector_type(4)));

__device__ __forceinline__ unsigned pcg(unsigned v) {
    v = v * 747796405u + 2891336453u;
    unsigned w = ((v >> ((v >> 28u) + 4u)) ^ v) * 277803737u;
    return (w >> 22u) ^ w;
}

// Detectron-legacy '+1' IoU. denom > 0 always (areas >= 1), rcp is safe.
__device__ __forceinline__ float iou_pair(float ax1, float ay1, float ax2, float ay2,
                                          const float* g) {
    float gx1 = g[0], gy1 = g[1], gx2 = g[2], gy2 = g[3];
    float aw = ax2 - ax1 + 1.f, ah = ay2 - ay1 + 1.f;
    float gw = gx2 - gx1 + 1.f, gh = gy2 - gy1 + 1.f;
    float ix = fminf(ax2, gx2) - fmaxf(ax1, gx1) + 1.f;
    float iy = fminf(ay2, gy2) - fmaxf(ay1, gy1) + 1.f;
    ix = fmaxf(ix, 0.f); iy = fmaxf(iy, 0.f);
    float inter = ix * iy;
    return inter * __builtin_amdgcn_rcpf(aw * ah + gw * gh - inter);
}

// Stage `nflt` contiguous f32 from global into LDS. Preferred path: CDNA5
// Tensor Data Mover (one DMA op, issued by wave 0, TENSORcnt-tracked).
// D# built per CDNA5 ISA: 1-D tile, data_size=4B, type=2.
// This toolchain exposes the 6-arg builtin:
//   (uint32x4 g0, int32x8 g1, int32x4 g2, int32x4 g3, int32x8 pad, i32 cpol)
// Caller must __syncthreads() afterwards.
__device__ __forceinline__ void stage_gt_lds(const float* __restrict__ src,
                                             float* dst, int nflt) {
#if __has_builtin(__builtin_amdgcn_tensor_load_to_lds)
    if ((threadIdx.x >> 5) == 0) {
        unsigned long long ga = (unsigned long long)(uintptr_t)src;
        unsigned la = (unsigned)(uintptr_t)dst;   // low 32 bits of flat shared = LDS offset
        unsigned n  = (unsigned)nflt;
        // group0: count=1 | lds_addr | global_addr[31:0] | global_addr[56:32] + type=2
        v4u g0 = { 1u, la, (unsigned)ga,
                   (unsigned)((ga >> 32) & 0x01FFFFFFull) | 0x80000000u };
        // group1: data_size=2 (4B); tensor_dim0=n @bits79:48; tensor_dim1=1 @111:80;
        // tile_dim0=n @127:112; tile_dim1=1 @143:128; dim0_stride=n @207:160;
        // dim1_stride=n @255:208
        v8i g1 = { (int)0x00020000u,
                   (int)((n & 0xFFFFu) << 16),
                   (int)((n >> 16) | (1u << 16)),
                   (int)((n & 0xFFFFu) << 16),
                   1,
                   (int)n,
                   (int)((n & 0xFFFFu) << 16),
                   (int)(n >> 16) };
        v4i z4 = { 0, 0, 0, 0 };
        v8i z8 = { 0, 0, 0, 0, 0, 0, 0, 0 };
        __builtin_amdgcn_tensor_load_to_lds(g0, g1, z4, z4, z8, 0);
        __builtin_amdgcn_s_wait_tensorcnt(0);
    }
#else
    for (int j = threadIdx.x; j < nflt; j += blockDim.x) dst[j] = src[j];
#endif
}

// ---------------- kernel 0: zero workspace counters -------------------
__global__ void k_init(unsigned* __restrict__ colmax, int* __restrict__ cnt) {
    int t = blockIdx.x * blockDim.x + threadIdx.x;
    if (t < N_B * N_G) colmax[t] = 0u;
    if (t < N_B)       cnt[t]    = 0;
}

// ---------------- kernel 1: per-(image,gt) column max -----------------
// Lane-owns-column layout: each lane privately accumulates max IoU for
// gt columns {lane, lane+32} over 32 anchors staged in LDS -> no shuffles.
// Block combines via LDS u32 atomic max (IoU >= 0 so bit order == float order),
// then one global atomic max per (b,g).
__global__ void k_pass1(const float* __restrict__ boxes, const float* __restrict__ gt,
                        const int* __restrict__ ign, const float* __restrict__ iminfo,
                        unsigned* __restrict__ colmax, int nA) {
    __shared__ float    sh_gt[N_B * N_G * 4];
    __shared__ float    sh_valid[N_B * N_G];
    __shared__ unsigned sh_cm[N_B * N_G];
    __shared__ float4   sh_box[256];
    int tid = threadIdx.x;
    stage_gt_lds(gt, sh_gt, N_B * N_G * 4);
    if (tid < N_B * N_G) {
        sh_valid[tid] = (ign[tid] == 0) ? 1.f : 0.f;
        sh_cm[tid] = 0u;
    }
    int ai = blockIdx.x * 256 + tid;
    if (ai >= nA) ai = nA - 1;                 // duplicate is harmless for max
    sh_box[tid] = ((const float4*)boxes)[ai];
    __syncthreads();

    float ih[N_B], iw[N_B];
#pragma unroll
    for (int b = 0; b < N_B; b++) { ih[b] = iminfo[b * 3]; iw[b] = iminfo[b * 3 + 1]; }

    int wave = tid >> 5, lane = tid & 31;
    float cm[N_B][2] = {};
    for (int a = 0; a < 32; a++) {
        float4 bb = sh_box[(wave << 5) + a];   // LDS broadcast read
#pragma unroll
        for (int b = 0; b < N_B; b++) {
            bool ins = (bb.x >= 0.f) && (bb.y >= 0.f) &&
                       (bb.z < iw[b] + 1.f) && (bb.w < ih[b] + 1.f);
#pragma unroll
            for (int t = 0; t < 2; t++) {
                int g = lane + (t << 5);
                float m = ins ? sh_valid[b * N_G + g] : 0.f;
                float u = iou_pair(bb.x, bb.y, bb.z, bb.w, &sh_gt[(b * N_G + g) * 4]) * m;
                cm[b][t] = fmaxf(cm[b][t], u);
            }
        }
    }
#pragma unroll
    for (int b = 0; b < N_B; b++)
#pragma unroll
        for (int t = 0; t < 2; t++)
            atomicMax(&sh_cm[b * N_G + lane + (t << 5)], __float_as_uint(cm[b][t]));
    __syncthreads();
    if (tid < N_B * N_G) atomicMax(&colmax[tid], sh_cm[tid]);
}

// ---------------- kernel 2: labels + provisional positives ------------
__global__ void k_pass2(const float* __restrict__ boxes, const float* __restrict__ gt,
                        const int* __restrict__ ign, const float* __restrict__ iminfo,
                        const unsigned* __restrict__ colmax,
                        int* __restrict__ labels, float* __restrict__ targets,
                        int* __restrict__ cnt, int* __restrict__ plist,
                        float* __restrict__ pprio, int nA) {
    __shared__ float sh_gt[N_B * N_G * 4];
    __shared__ float sh_valid[N_B * N_G];
    __shared__ float sh_cm[N_B * N_G];
    int tid = threadIdx.x;
    stage_gt_lds(gt, sh_gt, N_B * N_G * 4);
    if (tid < N_B * N_G) {
        sh_valid[tid] = (ign[tid] == 0) ? 1.f : 0.f;
        float c = __uint_as_float(colmax[tid]);
        sh_cm[tid] = (c == 0.f) ? 1.f : c;     // g2a_max==0 -> 1.0 fixup
    }
    __syncthreads();
    int i = blockIdx.x * blockDim.x + tid;
    if (i >= nA) return;
    float4 bb = ((const float4*)boxes)[i];
#pragma unroll
    for (int b = 0; b < N_B; b++) {
        float ih = iminfo[b * 3], iw = iminfo[b * 3 + 1];
        bool ins = (bb.x >= 0.f) && (bb.y >= 0.f) &&
                   (bb.z < iw + 1.f) && (bb.w < ih + 1.f);
        float amax = 0.f; int aidx = 0; bool wm = false;
        for (int g = 0; g < N_G; g++) {
            float m = ins ? sh_valid[b * N_G + g] : 0.f;
            float u = iou_pair(bb.x, bb.y, bb.z, bb.w, &sh_gt[(b * N_G + g) * 4]) * m;
            wm |= (u == sh_cm[b * N_G + g]);   // ensure_closest_box
            if (u > amax) { amax = u; aidx = g; }
        }
        bool pos = ins && (wm || amax >= POS_THR);
        long o = (long)b * nA + i;
        int lab = -1; float d0 = 0.f, d1 = 0.f, d2 = 0.f, d3 = 0.f;
        if (pos) {
            int slot = atomicAdd(&cnt[b], 1);
            if (slot < CAP) {
                lab = 1;
                plist[b * CAP + slot] = i;
                pprio[b * CAP + slot] =
                    (float)pcg(((unsigned)i << 2) ^ (unsigned)b) * (1.0f / 4294967296.0f);
                const float* g = &sh_gt[(b * N_G + aidx) * 4];
                float ew = bb.z - bb.x + 1.f, eh = bb.w - bb.y + 1.f;
                float ecx = bb.x + 0.5f * ew, ecy = bb.y + 0.5f * eh;
                float gw = g[2] - g[0] + 1.f, gh = g[3] - g[1] + 1.f;
                float gcx = g[0] + 0.5f * gw, gcy = g[1] + 0.5f * gh;
                d0 = (gcx - ecx) / ew; d1 = (gcy - ecy) / eh;
                d2 = __logf(gw / ew);  d3 = __logf(gh / eh);
            } // beyond CAP: demoted immediately (lab stays -1, deltas 0)
        }
        labels[o] = lab;
        targets[o * 4 + 0] = d0; targets[o * 4 + 1] = d1;
        targets[o * 4 + 2] = d2; targets[o * 4 + 3] = d3;
    }
}

// ---------------- kernel 3: subsample fg via WMMA rank, sample bg -----
// rank_i = sum_j [prio_i > prio_j]  ==  (0/1 indicator matrix) x ones-vector,
// computed on the matrix pipe with V_WMMA_F32_16X16X32_F16. Demote rank>=153.
__global__ void k_final(const float* __restrict__ boxes, const float* __restrict__ gt,
                        const int* __restrict__ ign, const float* __restrict__ iminfo,
                        int* __restrict__ labels, float* __restrict__ targets,
                        const int* __restrict__ cnt, const int* __restrict__ plist,
                        const float* __restrict__ pprio, int nA) {
    int b = blockIdx.x, tid = threadIdx.x;
    __shared__ float sh_gt[N_G * 4];
    __shared__ float sh_valid[N_G];
    __shared__ float sh_prio[CAP];
    __shared__ float sh_rank[CAP];
    for (int j = tid; j < N_G * 4; j += blockDim.x) sh_gt[j] = gt[b * N_G * 4 + j];
    if (tid < N_G) sh_valid[tid] = (ign[b * N_G + tid] == 0) ? 1.f : 0.f;
    int n = cnt[b]; if (n > CAP) n = CAP;
    int npad = (n + 31) & ~31;
    for (int j = tid; j < npad; j += blockDim.x)
        sh_prio[j] = (j < n) ? pprio[b * CAP + j] : 3.0e38f;  // pad = +big -> indicator 0
    __syncthreads();

    if (tid < 32 && n > 0) {                   // wave 0, EXEC all ones
        int lane = tid;
        v16h ones;
#pragma unroll
        for (int e = 0; e < 16; e++) ones[e] = (_Float16)1.0f;
        for (int m0 = 0; m0 < npad; m0 += 16) {
            float pm = sh_prio[m0 + (lane & 15)];   // row M = lane%16 of this tile
            v8f acc = {};
            for (int k0 = 0; k0 < npad; k0 += 32) {
                // 16-bit A layout: lanes 0-15 hold K in {0..7,16..23},
                // lanes 16-31 hold K in {8..15,24..31} of rows M=0..15.
                v16h amat;
#pragma unroll
                for (int e = 0; e < 16; e++) {
                    int k = k0 + ((lane < 16) ? ((e < 8) ? e : (e + 8))
                                              : ((e < 8) ? (e + 8) : (e + 16)));
                    amat[e] = (pm > sh_prio[k]) ? (_Float16)1.0f : (_Float16)0.0f;
                }
                acc = __builtin_amdgcn_wmma_f32_16x16x32_f16(
                    false, amat, false, ones, (short)0, acc, false, false);
            }
            // All D columns identical; lane0 holds M=0..7, lane16 holds M=8..15.
            if (lane == 0) {
#pragma unroll
                for (int r = 0; r < 8; r++) sh_rank[m0 + r] = acc[r];
            }
            if (lane == 16) {
#pragma unroll
                for (int r = 0; r < 8; r++) sh_rank[m0 + 8 + r] = acc[r];
            }
        }
    }
    __syncthreads();

    // demote positives ranked >= TARGET_FG
    for (int j = tid; j < n; j += blockDim.x) {
        if (sh_rank[j] > (float)TARGET_FG - 0.5f) {
            int idx = plist[b * CAP + j];
            long o = (long)b * nA + idx;
            labels[o] = -1;
            targets[o * 4 + 0] = 0.f; targets[o * 4 + 1] = 0.f;
            targets[o * 4 + 2] = 0.f; targets[o * 4 + 3] = 0.f;
        }
    }
    // sample negatives (uniform with replacement among neg anchors, rejection)
    int numfg = (n < TARGET_FG) ? n : TARGET_FG;
    int tbg = SAMPLE_BATCH - numfg;
    float ih = iminfo[b * 3], iw = iminfo[b * 3 + 1];
    const float4* bx = (const float4*)boxes;
    for (int s = tid; s < tbg; s += blockDim.x) {
        unsigned h = pcg(0x9E3779B9u ^ ((unsigned)b * 0x85EBCA6Bu) ^ ((unsigned)s * 0xC2B2AE35u));
        for (int att = 0; att < 8; att++) {
            int idx = (int)(h % (unsigned)nA);
            float4 bb = bx[idx];
            bool ins = (bb.x >= 0.f) && (bb.y >= 0.f) &&
                       (bb.z < iw + 1.f) && (bb.w < ih + 1.f);
            float amax = 0.f;
            if (ins) {
                for (int g = 0; g < N_G; g++) {
                    float u = iou_pair(bb.x, bb.y, bb.z, bb.w, &sh_gt[g * 4]) * sh_valid[g];
                    amax = fmaxf(amax, u);
                }
            }
            if (ins && amax < NEG_THR) { labels[(long)b * nA + idx] = 0; break; }
            h = pcg(h);
        }
    }
}

// ---------------- launch --------------------------------------------
extern "C" void kernel_launch(void* const* d_in, const int* in_sizes, int n_in,
                              void* d_out, int out_size, void* d_ws, size_t ws_size,
                              hipStream_t stream) {
    const float* boxes  = (const float*)d_in[0];
    const float* gt     = (const float*)d_in[1];
    // d_in[2] = gt_labels (unused, as in reference)
    const int*   ign    = (const int*)d_in[3];
    const float* iminfo = (const float*)d_in[4];
    int nA = in_sizes[0] / 4;

    int*   labels  = (int*)d_out;
    float* targets = (float*)d_out + (size_t)N_B * (size_t)nA;

    unsigned* ws_colmax = (unsigned*)d_ws;                                  // 1 KB
    int*      ws_cnt    = (int*)((char*)d_ws + 1024);                       // 16 B
    int*      ws_idx    = (int*)((char*)d_ws + 1088);                       // 16 KB
    float*    ws_prio   = (float*)((char*)d_ws + 1088 + sizeof(int) * N_B * CAP);

    int nBlocks = (nA + 255) / 256;
    k_init<<<1, 256, 0, stream>>>(ws_colmax, ws_cnt);
    k_pass1<<<nBlocks, 256, 0, stream>>>(boxes, gt, ign, iminfo, ws_colmax, nA);
    k_pass2<<<nBlocks, 256, 0, stream>>>(boxes, gt, ign, iminfo, ws_colmax,
                                         labels, targets, ws_cnt, ws_idx, ws_prio, nA);
    k_final<<<N_B, 256, 0, stream>>>(boxes, gt, ign, iminfo,
                                     labels, targets, ws_cnt, ws_idx, ws_prio, nA);
}
